// _Regression_Layer_86569360818764
// MI455X (gfx1250) — matrix operand order repaired and checked by reading, not compiled, not run on hardware
//
#include <hip/hip_runtime.h>
#include <hip/hip_bf16.h>

typedef __attribute__((ext_vector_type(16))) _Float16 v16h;
typedef __attribute__((ext_vector_type(8)))  float    v8f;

#define PP    7
#define HFD   14
#define CCH   256
#define TDIM  16
#define BDIM  4
#define RDIM  64
#define NROI  (BDIM * RDIM)          // 256
#define HW    (HFD * HFD)            // 196

// ---------------------------------------------------------------------------
// rois output: (B, R, T, 5)
// ---------------------------------------------------------------------------
__global__ void k_rois(const float* __restrict__ tubes, float* __restrict__ rois) {
  int idx = blockIdx.x * blockDim.x + threadIdx.x;
  if (idx >= BDIM * RDIM * TDIM) return;
  int t = idx % TDIM;
  int r = (idx / TDIM) % RDIM;
  int b = idx / (TDIM * RDIM);
  const float* tub = tubes + (size_t)(b * RDIM + r) * 7;
  int t1 = (int)tub[3], t2 = (int)tub[6];
  bool m = (t >= t1) && (t <= t2);
  float* o = rois + (size_t)idx * 5;
  o[0] = (float)(t + b * TDIM);
  o[1] = m ? tub[1] : 0.f;
  o[2] = m ? tub[2] : 0.f;
  o[3] = m ? tub[4] : 0.f;
  o[4] = m ? tub[5] : 0.f;
}

// ---------------------------------------------------------------------------
// inv channel norm per (b,t,h,w): 1 / max(||x||_C, 1e-12)
// ---------------------------------------------------------------------------
__global__ void k_invnorm(const float* __restrict__ bf, float* __restrict__ invn) {
  int p = blockIdx.x * blockDim.x + threadIdx.x;   // p = b*T*HW + t*HW + hw
  if (p >= BDIM * TDIM * HW) return;
  int b = p / (TDIM * HW);
  int rem = p % (TDIM * HW);
  const float* base = bf + (size_t)b * CCH * TDIM * HW + rem;
  float s = 0.f;
#pragma unroll 8
  for (int c = 0; c < CCH; ++c) {
    float v = base[(size_t)c * (TDIM * HW)];
    s += v * v;
  }
  invn[p] = 1.f / fmaxf(sqrtf(s), 1e-12f);
}

// ---------------------------------------------------------------------------
// avg over T of normalized features: avg[b,c,h,w] = (1/T) sum_t bf*invn
// ---------------------------------------------------------------------------
__global__ void k_avg(const float* __restrict__ bf, const float* __restrict__ invn,
                      float* __restrict__ avg) {
  int q = blockIdx.x * blockDim.x + threadIdx.x;   // q = (b*C + c)*HW + hw
  if (q >= BDIM * CCH * HW) return;
  int hw = q % HW;
  int c  = (q / HW) % CCH;
  int b  = q / (HW * CCH);
  const float* base = bf + ((size_t)(b * CCH + c) * TDIM) * HW + hw;
  const float* nb = invn + (size_t)b * TDIM * HW + hw;
  float acc = 0.f;
#pragma unroll
  for (int t = 0; t < TDIM; ++t) acc += base[t * HW] * nb[t * HW];
  avg[q] = acc * (1.f / (float)TDIM);
}

// ---------------------------------------------------------------------------
// ROI bilinear pooling on T-averaged features.
// grid = NROI blocks, 256 threads = channels. pooled: (N, C, 7, 7)
// ---------------------------------------------------------------------------
__global__ void k_pool(const float* __restrict__ tubes, const float* __restrict__ avg,
                       float* __restrict__ pooled) {
  int n = blockIdx.x;
  int c = threadIdx.x;
  const float* tub = tubes + (size_t)n * 7;
  int b = (int)tub[0];
  const float S = 1.f / 16.f;
  float x1 = tub[1] * S, y1 = tub[2] * S, x2 = tub[4] * S, y2 = tub[5] * S;
  float bw = fmaxf(x2 - x1 + S, 0.001f) * (1.f / (float)PP);
  float bh = fmaxf(y2 - y1 + S, 0.001f) * (1.f / (float)PP);
  const float* F = avg + (size_t)(b * CCH + c) * HW;
  float* outp = pooled + (size_t)(n * CCH + c) * (PP * PP);
#pragma unroll
  for (int cell = 0; cell < PP * PP; ++cell) {
    int ph = cell / PP, pw = cell % PP;
    float xs = fminf(fmaxf(x1 + (pw + 0.5f) * bw, 0.f), (float)(HFD - 1));
    float ys = fminf(fmaxf(y1 + (ph + 0.5f) * bh, 0.f), (float)(HFD - 1));
    float x0 = floorf(xs), y0 = floorf(ys);
    float fx = xs - x0, fy = ys - y0;
    int x0i = (int)x0, y0i = (int)y0;
    int x1i = min(x0i + 1, HFD - 1), y1i = min(y0i + 1, HFD - 1);
    float v00 = F[y0i * HFD + x0i], v01 = F[y0i * HFD + x1i];
    float v10 = F[y1i * HFD + x0i], v11 = F[y1i * HFD + x1i];
    outp[cell] = (1.f - fy) * ((1.f - fx) * v00 + fx * v01) +
                 fy * ((1.f - fx) * v10 + fx * v11);
  }
}

// ---------------------------------------------------------------------------
// Generic WMMA GEMM: C[m,n] = sum_k A[m,k]*B[k,n] (+ bias[n], optional relu)
// f32 inputs converted to f16 in-register, f32 WMMA accumulate.
// Compile-time element strides (enables immediate-offset global loads):
//   A[m,k] = A[m*lda + k*AKS + z*za]
//   B[k,n] = B[k*LDBK + n*BNS + z*zb]
//   C[m,n] = C[m*ldc + n*cns + z*zc]
// K is the per-block K extent; blockIdx.z selects either a spatial slice
// (conv) or a split-K chunk (SPLIT=true: raw partials, no bias/relu).
// Block = 128 threads = 4 waves (2x2); wave tile = 32x64 = 2x4 WMMA tiles
// (64 accumulator VGPRs + 48 fragment VGPRs -> no scratch spills).
// Requires M % 64 == 0, K % 32 == 0; N handled per-64 (waves skip OOB).
// ---------------------------------------------------------------------------
template <int AKS, int LDBK, int BNS, bool RELU, bool SPLIT>
__global__ void __launch_bounds__(128)
k_gemm(const float* __restrict__ A, const float* __restrict__ B,
       const float* __restrict__ bias, float* __restrict__ Cout,
       int M, int N, int K,
       long lda, long ldc, long cns, long za, long zb, long zc) {
  int lane  = threadIdx.x & 31;
  int w     = threadIdx.x >> 5;
  int row16 = lane & 15;
  int hi    = lane >> 4;          // 0: lanes 0-15, 1: lanes 16-31
  int m_base = blockIdx.y * 64  + (w >> 1) * 32;
  int n_base = blockIdx.x * 128 + (w & 1) * 64;
  if (m_base >= M || n_base >= N) return;   // wave-uniform

  long aoff = (long)blockIdx.z * za;
  long boff = (long)blockIdx.z * zb;
  long coff = (long)blockIdx.z * zc;

  // Per-lane base pointers (strength-reduced; all element accesses below use
  // compile-time immediate offsets).
  // A fragment: lane holds row m; halves i -> k = k0 + 16*hi + i
  //             (same contiguous order as B)
  const float* aptr0 = A + (long)(m_base +  0 + row16) * lda + aoff + (long)(hi * 16) * AKS;
  const float* aptr1 = A + (long)(m_base + 16 + row16) * lda + aoff + (long)(hi * 16) * AKS;
  // B fragment: lane holds col n; halves i -> k = k0 + 16*hi + i  (ISA B 32x16)
  const float* bptr0 = B + (long)(n_base +  0 + row16) * BNS + boff + (long)(hi * 16) * LDBK;
  const float* bptr1 = B + (long)(n_base + 16 + row16) * BNS + boff + (long)(hi * 16) * LDBK;
  const float* bptr2 = B + (long)(n_base + 32 + row16) * BNS + boff + (long)(hi * 16) * LDBK;
  const float* bptr3 = B + (long)(n_base + 48 + row16) * BNS + boff + (long)(hi * 16) * LDBK;

  v8f acc[2][4];
#pragma unroll
  for (int i = 0; i < 2; ++i)
#pragma unroll
    for (int j = 0; j < 4; ++j)
#pragma unroll
      for (int e = 0; e < 8; ++e) acc[i][j][e] = 0.f;

#pragma unroll 1
  for (int k0 = 0; k0 < K; k0 += 32) {
    if (LDBK > 1 && (k0 + 64 < K)) {
      __builtin_prefetch(bptr0 + (long)64 * LDBK, 0, 1);   // stream hint
    }
    v16h af[2], bfr[4];
#pragma unroll
    for (int i = 0; i < 8; ++i) {
      af[0][i]     = (_Float16)aptr0[i * AKS];
      af[0][8 + i] = (_Float16)aptr0[(8 + i) * AKS];
      af[1][i]     = (_Float16)aptr1[i * AKS];
      af[1][8 + i] = (_Float16)aptr1[(8 + i) * AKS];
    }
#pragma unroll
    for (int i = 0; i < 16; ++i) {
      bfr[0][i] = (_Float16)bptr0[i * LDBK];
      bfr[1][i] = (_Float16)bptr1[i * LDBK];
      bfr[2][i] = (_Float16)bptr2[i * LDBK];
      bfr[3][i] = (_Float16)bptr3[i * LDBK];
    }
#pragma unroll
    for (int tm = 0; tm < 2; ++tm)
#pragma unroll
      for (int tn = 0; tn < 4; ++tn)
        acc[tm][tn] = __builtin_amdgcn_wmma_f32_16x16x32_f16(
            false, af[tm], false, bfr[tn], (short)0, acc[tm][tn], false, false);

    aptr0 += 32 * AKS;  aptr1 += 32 * AKS;
    bptr0 += 32 * LDBK; bptr1 += 32 * LDBK;
    bptr2 += 32 * LDBK; bptr3 += 32 * LDBK;
  }

  // --- epilogue: bias (+relu) unless SPLIT partials. n = row16, m = 8*hi + r
#pragma unroll
  for (int tm = 0; tm < 2; ++tm)
#pragma unroll
    for (int tn = 0; tn < 4; ++tn) {
      int n = n_base + tn * 16 + row16;
      float bv = SPLIT ? 0.f : bias[n];
      long cb = (long)n * cns + coff;
#pragma unroll
      for (int r = 0; r < 8; ++r) {
        int m = m_base + tm * 16 + hi * 8 + r;
        float v = acc[tm][tn][r] + bv;
        if (RELU) v = fmaxf(v, 0.f);
        Cout[(long)m * ldc + cb] = v;
      }
    }
}

// ---------------------------------------------------------------------------
// split-K reduction: out = relu(sum_{s<4} part[s] + bias[n])
// ---------------------------------------------------------------------------
__global__ void k_reduce4(const float* __restrict__ part, const float* __restrict__ bias,
                          float* __restrict__ out, int total, int N) {
  int i = blockIdx.x * blockDim.x + threadIdx.x;
  if (i >= total) return;
  int n = i % N;
  float v = part[i] + part[i + (size_t)total] + part[i + 2 * (size_t)total] +
            part[i + 3 * (size_t)total] + bias[n];
  out[i] = fmaxf(v, 0.f);
}

// ---------------------------------------------------------------------------
extern "C" void kernel_launch(void* const* d_in, const int* in_sizes, int n_in,
                              void* d_out, int out_size, void* d_ws, size_t ws_size,
                              hipStream_t stream) {
  (void)in_sizes; (void)n_in; (void)out_size; (void)ws_size;
  const float* base_feat = (const float*)d_in[0];
  const float* tubes     = (const float*)d_in[1];
  // d_in[2] gt_rois, d_in[3] gt_tubes: unused by the forward computation
  const float* conv_w = (const float*)d_in[4];
  const float* conv_b = (const float*)d_in[5];
  const float* w1 = (const float*)d_in[6];
  const float* b1 = (const float*)d_in[7];
  const float* w2 = (const float*)d_in[8];
  const float* b2 = (const float*)d_in[9];
  const float* wp = (const float*)d_in[10];
  const float* bp = (const float*)d_in[11];

  float* out    = (float*)d_out;
  float* bbox   = out;                       // 256*64           = 16384
  float* pooled = out + 16384;               // 256*256*49       = 3211264
  float* rois   = out + 16384 + 3211264;     // 4*64*16*5        = 20480

  float* ws      = (float*)d_ws;
  float* invn    = ws;                  // 12544
  float* avg     = invn + 12544;        // 802816
  float* act1    = avg + 802816;        // 256*12544 = 3211264
  float* h1      = act1 + 3211264;      // 256*2048  = 524288
  float* h2      = h1 + 524288;         // 256*512   = 131072
  float* fc1part = h2 + 131072;         // 4 * 524288 = 2097152

  k_rois<<<(BDIM * RDIM * TDIM + 255) / 256, 256, 0, stream>>>(tubes, rois);
  k_invnorm<<<(BDIM * TDIM * HW + 255) / 256, 256, 0, stream>>>(base_feat, invn);
  k_avg<<<(BDIM * CCH * HW + 255) / 256, 256, 0, stream>>>(base_feat, invn, avg);
  k_pool<<<NROI, CCH, 0, stream>>>(tubes, avg, pooled);

  dim3 blk(128, 1, 1);
  // 1x1 conv as 49 GEMMs (z = spatial cell): M=256(n), N=256(o), K=256(c)
  //   A[m,k] = pooled[m*12544 + k*49 + z];  B[k,n] = conv_w[n*256 + k]
  //   C[m,n] = act1[m*12544 + n*49 + z]
  k_gemm<49, 1, 256, false, false><<<dim3(2, 4, 49), blk, 0, stream>>>(
      pooled, conv_w, conv_b, act1, 256, 256, 256,
      /*lda*/ 12544, /*ldc*/ 12544, /*cns*/ 49, /*za*/ 1, /*zb*/ 0, /*zc*/ 1);
  // FC1: 256 x 2048 x 12544, split-K x4 (chunks of 3136) -> partials, then
  // fused reduce + bias + relu.
  k_gemm<1, 2048, 1, false, true><<<dim3(16, 4, 4), blk, 0, stream>>>(
      act1, w1, b1, fc1part, 256, 2048, /*Kchunk*/ 3136,
      /*lda*/ 12544, /*ldc*/ 2048, /*cns*/ 1,
      /*za*/ 3136, /*zb*/ 3136L * 2048, /*zc*/ 256L * 2048);
  k_reduce4<<<(524288 + 255) / 256, 256, 0, stream>>>(fc1part, b1, h1, 524288, 2048);
  // FC2: 256 x 512 x 2048, relu
  k_gemm<1, 512, 1, true, false><<<dim3(4, 4, 1), blk, 0, stream>>>(
      h1, w2, b2, h2, 256, 512, 2048,
      2048, 512, 1, 0, 0, 0);
  // FC3: 256 x 64 x 512 -> bbox_pred
  k_gemm<1, 64, 1, false, false><<<dim3(1, 4, 1), blk, 0, stream>>>(
      h2, wp, bp, bbox, 256, 64, 512,
      512, 64, 1, 0, 0, 0);
}
